// TRACELoss_43370579755214
// MI455X (gfx1250) — compile-verified, matrix operand order checked by reading, not run on hardware
//
#include <hip/hip_runtime.h>
#include <stdint.h>

typedef unsigned int u32;

#define NCH   16
#define HISTB 256

// ws layout in u32 words
#define OFF_NUMPOS 0
#define OFF_SPOS   16
#define OFF_PREFIX 32
#define OFF_KREM   48
#define OFF_SABOVE 64
#define OFF_NUMNEG 80
#define OFF_HCNT   256
#define OFF_HSUM   (OFF_HCNT + NCH*HISTB)           // 256+4096 = 4352
#define OFF_CACHE  (OFF_HSUM + NCH*HISTB + 256)     // 8704 words (16B aligned)
#define STATS_WORDS OFF_CACHE

// huber(delta=1), w*hub, lt bits. NOTE: relies on lt >= 0 (w in [0,1), hub >= 0),
// so float ordering == uint-bit ordering for the radix select.
__device__ __forceinline__ u32 lt_bits_one(float p, float g, float w,
                                           bool& is_pos, float& wh) {
  float d  = p - g;
  float ad = fabsf(d);
  float hub = ad < 1.0f ? 0.5f * d * d : ad - 0.5f;
  wh = w * hub;
  is_pos = g > 0.0f;
  float lt = is_pos ? 0.0f : wh;
  return __float_as_uint(lt);
}

__global__ void __launch_bounds__(256)
k_zero(u32* ws) {
  for (int i = threadIdx.x; i < STATS_WORDS; i += 256) ws[i] = 0;
}

// Pass 1: per-channel pos count/sum + level-0 histogram (top byte of lt bits),
// with per-bin count and per-bin value sum. Optionally caches lt bits in ws.
__global__ void __launch_bounds__(256)
k_scan(const float4* __restrict__ p4, const float4* __restrict__ g4,
       const float4* __restrict__ w4, u32* __restrict__ ws,
       uint4* __restrict__ cache, int npos, int use_cache) {
  __shared__ u32   hcnt[NCH * HISTB];
  __shared__ float hsum[NCH * HISTB];
  __shared__ u32   pcnt_s[NCH];
  __shared__ float psum_s[NCH];
  for (int i = threadIdx.x; i < NCH * HISTB; i += blockDim.x) { hcnt[i] = 0; hsum[i] = 0.0f; }
  if (threadIdx.x < NCH) { pcnt_s[threadIdx.x] = 0; psum_s[threadIdx.x] = 0.0f; }
  __syncthreads();

  u32 my_pc[NCH]; float my_ps[NCH]; u32 my_z[NCH];
#pragma unroll
  for (int c = 0; c < NCH; ++c) { my_pc[c] = 0; my_ps[c] = 0.0f; my_z[c] = 0; }

  int stride = blockDim.x * gridDim.x;
  for (int pos = blockIdx.x * blockDim.x + threadIdx.x; pos < npos; pos += stride) {
    if (pos + stride < npos) {  // speculative streaming prefetch (global_prefetch_b8)
      __builtin_prefetch(&p4[(size_t)(pos + stride) * 4], 0, 0);
      __builtin_prefetch(&g4[(size_t)(pos + stride) * 4], 0, 0);
      __builtin_prefetch(&w4[(size_t)(pos + stride) * 4], 0, 0);
    }
#pragma unroll
    for (int j = 0; j < 4; ++j) {
      float4 pv = p4[(size_t)pos * 4 + j];
      float4 gv = g4[(size_t)pos * 4 + j];
      float4 wv = w4[(size_t)pos * 4 + j];
      float pp[4] = {pv.x, pv.y, pv.z, pv.w};
      float gg[4] = {gv.x, gv.y, gv.z, gv.w};
      float ww[4] = {wv.x, wv.y, wv.z, wv.w};
      u32 bits4[4];
#pragma unroll
      for (int e = 0; e < 4; ++e) {
        int c = j * 4 + e;
        bool is_pos; float wh;
        u32 bits = lt_bits_one(pp[e], gg[e], ww[e], is_pos, wh);
        bits4[e] = bits;
        if (is_pos) { my_pc[c]++; my_ps[c] += wh; }
        if (bits == 0u) {
          my_z[c]++;  // exact zeros: batch into bin 0 later (kills hot-bin contention)
        } else {
          int bin = bits >> 24;
          atomicAdd(&hcnt[c * HISTB + bin], 1u);
          atomicAdd(&hsum[c * HISTB + bin], __uint_as_float(bits));
        }
      }
      if (use_cache) cache[(size_t)pos * 4 + j] = make_uint4(bits4[0], bits4[1], bits4[2], bits4[3]);
    }
  }
#pragma unroll
  for (int c = 0; c < NCH; ++c) {
    if (my_z[c]) atomicAdd(&hcnt[c * HISTB + 0], my_z[c]);
    atomicAdd(&pcnt_s[c], my_pc[c]);
    atomicAdd(&psum_s[c], my_ps[c]);
  }
  __syncthreads();
  if (threadIdx.x < NCH) {
    atomicAdd(&ws[OFF_NUMPOS + threadIdx.x], pcnt_s[threadIdx.x]);
    atomicAdd((float*)ws + OFF_SPOS + threadIdx.x, psum_s[threadIdx.x]);
  }
  for (int i = threadIdx.x; i < NCH * HISTB; i += blockDim.x) {
    u32 cn = hcnt[i];
    if (cn) atomicAdd(&ws[OFF_HCNT + i], cn);
    float s = hsum[i];
    if (s != 0.0f) atomicAdd((float*)ws + OFF_HSUM + i, s);
  }
}

// Per-level bin selection: walk bins high->low, accumulate count/sum above the
// k-th largest, update prefix/k_rem, then zero the histogram for the next level.
__global__ void __launch_bounds__(256)
k_select(u32* ws, int npos, int level) {
  int t = threadIdx.x;
  if (t < NCH) {
    u32 krem;
    if (level == 0) {
      u32 np = ws[OFF_NUMPOS + t];
      u32 nneg;
      if (np > 0) {
        nneg = 3u * np;
        u32 cap = (u32)(npos - 1);
        if (nneg > cap) nneg = cap;
      } else {
        nneg = 10000u;  // DEFAULT_NUM_NEG
      }
      ws[OFF_NUMNEG + t] = nneg;
      ws[OFF_PREFIX + t] = 0u;
      ((float*)ws)[OFF_SABOVE + t] = 0.0f;
      krem = nneg;
    } else {
      krem = ws[OFF_KREM + t];
    }
    u32*   hc = ws + OFF_HCNT + t * HISTB;
    float* hs = (float*)ws + OFF_HSUM + t * HISTB;
    u32 cnt_above = 0; float s_above = 0.0f; int bsel = 0;
    for (int b = HISTB - 1; b >= 0; --b) {
      u32 cb = hc[b];
      if (cnt_above + cb >= krem) { bsel = b; break; }
      cnt_above += cb; s_above += hs[b];
    }
    ws[OFF_PREFIX + t] |= (u32)bsel << (24 - 8 * level);
    ws[OFF_KREM + t]    = krem - cnt_above;
    ((float*)ws)[OFF_SABOVE + t] += s_above;
  }
  __syncthreads();
  for (int i = threadIdx.x; i < 2 * NCH * HISTB; i += blockDim.x) ws[OFF_HCNT + i] = 0;
}

// Refine pass (level 1..3): histogram next byte for elements matching the prefix.
__global__ void __launch_bounds__(256)
k_refine(const float4* __restrict__ p4, const float4* __restrict__ g4,
         const float4* __restrict__ w4, u32* __restrict__ ws,
         const uint4* __restrict__ cache, int npos, int level, int use_cache) {
  __shared__ u32   hcnt[NCH * HISTB];
  __shared__ float hsum[NCH * HISTB];
  for (int i = threadIdx.x; i < NCH * HISTB; i += blockDim.x) { hcnt[i] = 0; hsum[i] = 0.0f; }
  __syncthreads();

  u32 pref[NCH];
#pragma unroll
  for (int c = 0; c < NCH; ++c) pref[c] = ws[OFF_PREFIX + c];  // uniform -> SGPR loads
  u32 mask  = 0xFFFFFFFFu << (32 - 8 * level);
  int shift = 24 - 8 * level;

  int stride = blockDim.x * gridDim.x;
  for (int pos = blockIdx.x * blockDim.x + threadIdx.x; pos < npos; pos += stride) {
#pragma unroll
    for (int j = 0; j < 4; ++j) {
      u32 bits4[4];
      if (use_cache) {
        uint4 v = cache[(size_t)pos * 4 + j];
        bits4[0] = v.x; bits4[1] = v.y; bits4[2] = v.z; bits4[3] = v.w;
      } else {
        float4 pv = p4[(size_t)pos * 4 + j];
        float4 gv = g4[(size_t)pos * 4 + j];
        float4 wv = w4[(size_t)pos * 4 + j];
        float pp[4] = {pv.x, pv.y, pv.z, pv.w};
        float gg[4] = {gv.x, gv.y, gv.z, gv.w};
        float ww[4] = {wv.x, wv.y, wv.z, wv.w};
#pragma unroll
        for (int e = 0; e < 4; ++e) { bool ip; float wh; bits4[e] = lt_bits_one(pp[e], gg[e], ww[e], ip, wh); }
      }
#pragma unroll
      for (int e = 0; e < 4; ++e) {
        int c = j * 4 + e;
        u32 bits = bits4[e];
        if ((bits & mask) == pref[c]) {
          int bin = (bits >> shift) & 255;
          atomicAdd(&hcnt[c * HISTB + bin], 1u);
          atomicAdd(&hsum[c * HISTB + bin], __uint_as_float(bits));
        }
      }
    }
  }
  __syncthreads();
  for (int i = threadIdx.x; i < NCH * HISTB; i += blockDim.x) {
    u32 cn = hcnt[i];
    if (cn) atomicAdd(&ws[OFF_HCNT + i], cn);
    float s = hsum[i];
    if (s != 0.0f) atomicAdd((float*)ws + OFF_HSUM + i, s);
  }
}

typedef __attribute__((ext_vector_type(2)))  float    v2f;
typedef __attribute__((ext_vector_type(8)))  float    v8f;
typedef __attribute__((ext_vector_type(16))) _Float16 v16h;

// Finalize: per-channel loss in lanes 0..15, reduce across channels with WMMA.
__global__ void __launch_bounds__(32)
k_final(const u32* __restrict__ ws, float* __restrict__ out) {
  int l = threadIdx.x;  // 0..31, one wave32
  float Lc = 0.0f;
  if (l < NCH) {
    u32 np = ws[OFF_NUMPOS + l];
    u32 nn = ws[OFF_NUMNEG + l];
    float sp = ((const float*)ws)[OFF_SPOS + l];
    float sa = ((const float*)ws)[OFF_SABOVE + l];
    u32 kr = ws[OFF_KREM + l];
    float tv = __uint_as_float(ws[OFF_PREFIX + l]);  // exact k-th largest lt value
    Lc = (sp + sa + (float)kr * tv) / (float)(np + nn);
  }
#if __has_builtin(__builtin_amdgcn_wmma_f32_16x16x4_f32)
  // A(16x4 f32): lanes 0-15 hold {K0,K1}, lanes 16-31 hold {K2,K3} for row M=lane%16.
  // Pack A[m][k] = L[4m+k] (m<4), B = ones -> D[m][*] = sum_k L[4m+k].
  int m     = (l < 16) ? l : (l - 16);
  int kbase = (l < 16) ? 0 : 2;
  float ax = __shfl(Lc, 4 * m + kbase,     32);
  float ay = __shfl(Lc, 4 * m + kbase + 1, 32);
  bool act = m < 4;
  v2f a; a.x = act ? ax : 0.0f; a.y = act ? ay : 0.0f;
  v2f b; b.x = 1.0f; b.y = 1.0f;
  v8f c = {};
  v8f d = __builtin_amdgcn_wmma_f32_16x16x4_f32(false, a, false, b, (short)0, c, false, false);
  if (l == 0) out[0] = d[0] + d[1] + d[2] + d[3] + d[4] + d[5] + d[6] + d[7];
#else
  // Fallback: f16 WMMA. Lane 0 holds L[0..7] (K0-7) and L[8..15] (K16-23) of row M=0.
  v16h a = {};
  if (l == 0) {
#pragma unroll
    for (int i = 0; i < 16; ++i) a[i] = (_Float16)__shfl(Lc, i, 32);
  }
  v16h b;
#pragma unroll
  for (int i = 0; i < 16; ++i) b[i] = (_Float16)1.0f;
  v8f c = {};
  v8f d = __builtin_amdgcn_wmma_f32_16x16x32_f16(false, a, false, b, (short)0, c, false, false);
  if (l == 0) out[0] = d[0];
#endif
}

extern "C" void kernel_launch(void* const* d_in, const int* in_sizes, int n_in,
                              void* d_out, int out_size, void* d_ws, size_t ws_size,
                              hipStream_t stream) {
  const float* p = (const float*)d_in[0];
  const float* g = (const float*)d_in[1];
  const float* w = (const float*)d_in[2];
  int total = in_sizes[0];
  int npos  = total / NCH;  // per-channel length N

  u32* ws = (u32*)d_ws;
  size_t need_cache = (size_t)STATS_WORDS * 4 + (size_t)npos * NCH * 4;
  int use_cache = (ws_size >= need_cache) ? 1 : 0;
  uint4* cache = (uint4*)(ws + OFF_CACHE);

  const int BLK = 256, GRD = 512;
  k_zero<<<1, BLK, 0, stream>>>(ws);
  k_scan<<<GRD, BLK, 0, stream>>>((const float4*)p, (const float4*)g, (const float4*)w,
                                  ws, cache, npos, use_cache);
  k_select<<<1, BLK, 0, stream>>>(ws, npos, 0);
  for (int lvl = 1; lvl <= 3; ++lvl) {
    k_refine<<<GRD, BLK, 0, stream>>>((const float4*)p, (const float4*)g, (const float4*)w,
                                      ws, cache, npos, lvl, use_cache);
    k_select<<<1, BLK, 0, stream>>>(ws, npos, lvl);
  }
  k_final<<<1, 32, 0, stream>>>(ws, (float*)d_out);
}
